// LayoutLMForSegmentation_41360535060987
// MI455X (gfx1250) — matrix operand order repaired and checked by reading, not compiled, not run on hardware
//
#include <hip/hip_runtime.h>
#include <hip/hip_bf16.h>

#define BATCH 2
#define SEQ   512
#define HID   768
#define FEAT  256
#define HH    512
#define WW    512
#define NCLS  9

#if defined(__AMDGCN__) && __has_builtin(__builtin_amdgcn_global_load_async_to_lds_b128)
#define HAVE_ASYNC_LDS 1
#else
#define HAVE_ASYNC_LDS 0
#endif

typedef __attribute__((ext_vector_type(16))) __bf16 v16bf;
typedef __attribute__((ext_vector_type(8)))  __bf16 v8bf;
typedef __attribute__((ext_vector_type(8)))  float  v8f;
typedef __attribute__((__vector_size__(4 * sizeof(int)))) int v4i;

static __device__ __forceinline__ __bf16 f2bf(float f) {
  union { float f; unsigned u; } x; x.f = f;
  unsigned r = (x.u + 0x7FFFu + ((x.u >> 16) & 1u)) >> 16;
  union { unsigned short s; __bf16 b; } o; o.s = (unsigned short)r;
  return o.b;
}
static __device__ __forceinline__ int imax(int a, int b) { return a > b ? a : b; }
static __device__ __forceinline__ int imin(int a, int b) { return a < b ? a : b; }

// ---------------------------------------------------------------------------
// Weight repack: source [K][N] row-major (projection weight) -> WMMA B wave
// tiles. Packed element layout: ((kt*NT + nt)*32 + lane)*16 + j, where the
// bf16 at slot j of lane l is B[kt*32 + (j/8)*16 + (l/16)*8 + j%8][nt*16 + (l%16)].
// ---------------------------------------------------------------------------
__global__ void pack_kn(const float* __restrict__ w, __bf16* __restrict__ p,
                        int K, int N) {
  int i = blockIdx.x * blockDim.x + threadIdx.x;
  if (i >= K * N) return;
  int j = i & 15, lane = (i >> 4) & 31, tile = i >> 9;
  int NT = N >> 4;
  int kt = tile / NT, nt = tile - kt * NT;
  int k = kt * 32 + ((j & 8) << 1) + ((lane >> 4) << 3) + (j & 7);
  int n = nt * 16 + (lane & 15);
  p[i] = f2bf(w[(size_t)k * N + n]);
}

// Conv weight repack: OIHW fp32 -> WMMA B wave tiles with GEMM K = tap*C + c,
// N = out-channel (zero-padded to NPAD).
__global__ void pack_oihw(const float* __restrict__ w, __bf16* __restrict__ p,
                          int O, int C, int KHW, int NPAD) {
  int i = blockIdx.x * blockDim.x + threadIdx.x;
  int total = ((C * KHW) >> 5) * (NPAD >> 4) * 512;
  if (i >= total) return;
  int j = i & 15, lane = (i >> 4) & 31, tile = i >> 9;
  int NT = NPAD >> 4;
  int kt = tile / NT, nt = tile - kt * NT;
  int k = kt * 32 + ((j & 8) << 1) + ((lane >> 4) << 3) + (j & 7);
  int n = nt * 16 + (lane & 15);
  int tap = k / C, c = k - tap * C;
  float v = (n < O) ? w[((size_t)n * C + c) * KHW + tap] : 0.0f;
  p[i] = f2bf(v);
}

// ---------------------------------------------------------------------------
// Projection GEMM: features[1024,256] = hs[1024,768] @ proj_w + b, bf16 out.
// ---------------------------------------------------------------------------
__global__ void __launch_bounds__(256) proj_wmma(
    const float* __restrict__ hs, const __bf16* __restrict__ wpk,
    const float* __restrict__ bias, __bf16* __restrict__ feat) {
  const int lane = threadIdx.x & 31;
  const int wid  = blockIdx.x * 8 + (threadIdx.x >> 5);
  if (wid >= (BATCH * SEQ / 16) * (FEAT / 16)) return;
  const int mt = wid >> 4;
  const int nt = wid & 15;
  const int kb = (lane & 16) ? 8 : 0;
  const int ml = lane & 15;
  const float* arow = hs + (size_t)(mt * 16 + ml) * HID + kb;

  v8f c;
  #pragma unroll
  for (int r = 0; r < 8; ++r) c[r] = 0.0f;

  for (int k0 = 0; k0 < HID; k0 += 32) {
    v16bf a;
    #pragma unroll
    for (int j = 0; j < 8; ++j) {
      a[j]     = f2bf(arow[k0 + j]);
      a[j + 8] = f2bf(arow[k0 + 16 + j]);
    }
    const v16bf bm = *(const v16bf*)(wpk +
        ((size_t)((k0 >> 5) * (FEAT / 16) + nt) * 32 + lane) * 16);
    c = __builtin_amdgcn_wmma_f32_16x16x32_bf16(false, a, false, bm,
                                                (short)0, c, false, false);
  }
  const int n    = nt * 16 + ml;
  const float bi = bias[n];
  const int mrow = mt * 16 + ((lane >> 4) << 3);
  #pragma unroll
  for (int r = 0; r < 8; ++r)
    feat[(size_t)(mrow + r) * FEAT + n] = f2bf(c[r] + bi);
}

// ---------------------------------------------------------------------------
// Rasterization: sequential overwrite == per-pixel max token index.
// ---------------------------------------------------------------------------
__global__ void init_winner(int* __restrict__ w, int n) {
  int i = blockIdx.x * blockDim.x + threadIdx.x;
  if (i < n) w[i] = 0;
}

__global__ void __launch_bounds__(256) rasterize(
    const int* __restrict__ bbox, const int* __restrict__ amask,
    int* __restrict__ winner) {
  int tok  = blockIdx.x * 8 + (threadIdx.x >> 5);
  int lane = threadIdx.x & 31;
  int b = tok / SEQ, s = tok - b * SEQ;
  if (s == 0) return;
  if (amask[tok] != 1) return;
  const int* bb = bbox + (size_t)tok * 4;
  int x1 = imax(bb[0] * WW / 1000, 0);
  int y1 = imax(bb[1] * HH / 1000, 0);
  int x2 = imin(bb[2] * WW / 1000, WW - 1);
  int y2 = imin(bb[3] * HH / 1000, HH - 1);
  if (x2 <= x1 || y2 <= y1) return;
  int bw = x2 - x1, bh = y2 - y1;
  int total = bw * bh;
  for (int i = lane; i < total; i += 32) {
    int yy = y1 + i / bw;
    int xx = x1 + (i - (i / bw) * bw);
    atomicMax(&winner[(size_t)(b * HH + yy) * WW + xx], s);
  }
}

// Gather NHWC bf16 feature map: one wave per pixel, 8 channels/lane.
__global__ void __launch_bounds__(256) gather_fmap(
    const int* __restrict__ winner, const __bf16* __restrict__ feat,
    __bf16* __restrict__ fmap) {
  int pix  = blockIdx.x * 8 + (threadIdx.x >> 5);
  int lane = threadIdx.x & 31;
  int idx  = winner[pix];
  int b    = pix / (HH * WW);
  union { v8bf v; unsigned long long q[2]; } val;
  if (idx > 0) {
    val.v = *(const v8bf*)(feat + (size_t)(b * SEQ + idx) * FEAT + lane * 8);
  } else {
    val.q[0] = 0ull; val.q[1] = 0ull;
  }
  *(v8bf*)(fmap + (size_t)pix * FEAT + lane * 8) = val.v;
}

// ---------------------------------------------------------------------------
// 3x3 conv, implicit GEMM with double-buffered LDS staging + fused BN+ReLU.
// Block (8 waves) owns a 16x16 output pixel tile. Input channels are staged
// in chunks of CCH=64 into 18x18 halo tiles (2 x 40.5 KB LDS). Staging uses
// CDNA5 async global->LDS DMA (ASYNCcnt) so the next chunk streams in while
// WMMA consumes the current one. Each wave owns 2*NG jobs = (output row,
// 64-out-channel group); accumulators live across the whole K loop.
// ---------------------------------------------------------------------------
template <int CIN, int NG>
__global__ void __launch_bounds__(256) conv3x3_lds(
    const __bf16* __restrict__ in, const __bf16* __restrict__ wpk,
    const float* __restrict__ bias, const float* __restrict__ bng,
    const float* __restrict__ bnb, const float* __restrict__ bnm,
    const float* __restrict__ bnv, __bf16* __restrict__ outb) {
  constexpr int CCH  = 64;            // channel chunk staged in LDS
  constexpr int NWT  = 4;             // 4 n-tiles (64 out channels) per job
  constexpr int NJ   = 2 * NG;        // jobs per wave
  constexpr int COUT = NG * 64;
  constexpr int NT   = NG * 4;        // n-tiles total in packed weights

  __shared__ __bf16 smem[2][18 * 18 * CCH];

  const int tid  = threadIdx.x;
  const int lane = tid & 31;
  const int wv   = tid >> 5;
  const int kb   = (lane & 16) ? 8 : 0;
  const int ml   = lane & 15;

  const int bid = blockIdx.x;                  // BATCH*(HH/16)*(WW/16)
  const int b   = bid / ((HH / 16) * (WW / 16));
  const int rem = bid - b * ((HH / 16) * (WW / 16));
  const int y0  = (rem / (WW / 16)) * 16;
  const int x0  = (rem - (rem / (WW / 16)) * (WW / 16)) * 16;

  // Stage one CCH-channel chunk of the 18x18 halo tile into smem[buf].
  auto stage = [&](int buf, int cc) {
    for (int s = tid; s < 18 * 18 * (CCH / 8); s += 256) {
      const int c8  = s & (CCH / 8 - 1);
      const int pxl = s >> 3;                 // CCH/8 == 8
      const int ry  = pxl / 18;
      const int rx  = pxl - ry * 18;
      const int gy  = y0 + ry - 1;
      const int gx  = x0 + rx - 1;
      __bf16* dst = &smem[buf][(size_t)pxl * CCH + c8 * 8];
      if (gy >= 0 && gy < HH && gx >= 0 && gx < WW) {
        const __bf16* src = in +
            ((size_t)(b * HH + gy) * WW + gx) * CIN + cc + c8 * 8;
#if HAVE_ASYNC_LDS
        __builtin_amdgcn_global_load_async_to_lds_b128(
            (v4i*)src, (v4i*)dst, 0, 0);
#else
        *(v8bf*)dst = *(const v8bf*)src;
#endif
      } else {
        union { v8bf v; unsigned long long q[2]; } z;
        z.q[0] = 0ull; z.q[1] = 0ull;
        *(v8bf*)dst = z.v;
      }
    }
  };

  v8f acc[NJ * NWT];
  #pragma unroll
  for (int t = 0; t < NJ * NWT; ++t)
    #pragma unroll
    for (int r = 0; r < 8; ++r) acc[t][r] = 0.0f;

  stage(0, 0);                                  // prologue: chunk 0 in flight
  int pb = 0;
  for (int cc = 0; cc < CIN; cc += CCH, pb ^= 1) {
#if HAVE_ASYNC_LDS
    asm volatile("s_wait_asynccnt 0x0" ::: "memory");
#endif
    __syncthreads();                            // smem[pb] ready for everyone
    if (cc + CCH < CIN) stage(pb ^ 1, cc + CCH);  // overlap next chunk DMA

    for (int tap = 0; tap < 9; ++tap) {
      const int dy = tap / 3 - 1;
      const int dx = tap - (tap / 3) * 3 - 1;
      #pragma unroll
      for (int j = 0; j < NJ; ++j) {
        const int jid = wv * NJ + j;
        const int row = jid / NG;
        const int ng  = jid - row * NG;
        const int ry  = row + dy + 1;
        const int rx  = ml + dx + 1;
        const __bf16* ap = &smem[pb][(ry * 18 + rx) * CCH + kb];
        for (int c0 = 0; c0 < CCH; c0 += 32) {
          union { v16bf v; v8bf h[2]; } a;
          a.h[0] = *(const v8bf*)(ap + c0);
          a.h[1] = *(const v8bf*)(ap + c0 + 16);
          const int kt = (tap * CIN + cc + c0) >> 5;
          #pragma unroll
          for (int t = 0; t < NWT; ++t) {
            const v16bf bm = *(const v16bf*)(wpk +
                ((size_t)(kt * NT + ng * NWT + t) * 32 + lane) * 16);
            acc[j * NWT + t] = __builtin_amdgcn_wmma_f32_16x16x32_bf16(
                false, a.v, false, bm, (short)0, acc[j * NWT + t],
                false, false);
          }
        }
      }
    }
  }

  // ---- epilogue: fused bias + BN + ReLU, bf16 NHWC store ----
  const int mb = (lane >> 4) << 3;
  #pragma unroll
  for (int j = 0; j < NJ; ++j) {
    const int jid = wv * NJ + j;
    const int row = jid / NG;
    const int ng  = jid - row * NG;
    const int y   = y0 + row;
    const size_t pix = (size_t)(b * HH + y) * WW + x0 + mb;
    #pragma unroll
    for (int t = 0; t < NWT; ++t) {
      const int ch  = (ng * NWT + t) * 16 + ml;
      const float s  = bng[ch] / sqrtf(bnv[ch] + 1e-5f);
      const float sh = bnb[ch] + (bias[ch] - bnm[ch]) * s;
      #pragma unroll
      for (int r = 0; r < 8; ++r) {
        float vv = fmaxf(acc[j * NWT + t][r] * s + sh, 0.0f);
        outb[(pix + r) * COUT + ch] = f2bf(vv);
      }
    }
  }
}

// ---------------------------------------------------------------------------
// Final 1x1 conv (64 -> 9), direct implicit GEMM, f32 NCHW logits.
// ---------------------------------------------------------------------------
__global__ void __launch_bounds__(256) conv1x1_final(
    const __bf16* __restrict__ in, const __bf16* __restrict__ wpk,
    const float* __restrict__ bias, float* __restrict__ outf) {
  constexpr int CIN = 64;
  const int lane = threadIdx.x & 31;
  const int gid  = blockIdx.x * 8 + (threadIdx.x >> 5);
  const int b    = gid / (HH * (WW / 16));
  int rem        = gid - b * (HH * (WW / 16));
  const int y    = rem / (WW / 16);
  const int x0   = (rem - y * (WW / 16)) * 16;
  const int kb   = (lane & 16) ? 8 : 0;
  const int ml   = lane & 15;

  v8f acc;
  #pragma unroll
  for (int r = 0; r < 8; ++r) acc[r] = 0.0f;

  const __bf16* arow = in + ((size_t)(b * HH + y) * WW + x0 + ml) * CIN + kb;
  #pragma unroll
  for (int c0 = 0; c0 < CIN; c0 += 32) {
    union { v16bf v; v8bf h[2]; } a;
    a.h[0] = *(const v8bf*)(arow + c0);
    a.h[1] = *(const v8bf*)(arow + c0 + 16);
    const v16bf bm = *(const v16bf*)(wpk + ((size_t)(c0 >> 5) * 32 + lane) * 16);
    acc = __builtin_amdgcn_wmma_f32_16x16x32_bf16(false, a.v, false, bm,
                                                  (short)0, acc, false, false);
  }

  const int mb = (lane >> 4) << 3;
  const int ch = ml;
  if (ch < NCLS) {
    const float bi = bias[ch];
    #pragma unroll
    for (int r = 0; r < 8; ++r)
      outf[((size_t)(b * NCLS + ch) * HH + y) * WW + x0 + mb + r] = acc[r] + bi;
  }
}

// ---------------------------------------------------------------------------
extern "C" void kernel_launch(void* const* d_in, const int* in_sizes, int n_in,
                              void* d_out, int out_size, void* d_ws, size_t ws_size,
                              hipStream_t stream) {
  const float* hs    = (const float*)d_in[0];
  const int*   bbox  = (const int*)d_in[1];
  const int*   amask = (const int*)d_in[2];
  const float* pw    = (const float*)d_in[3];
  const float* pb    = (const float*)d_in[4];
  const float* w1    = (const float*)d_in[5];
  const float* b1    = (const float*)d_in[6];
  const float* g1 = (const float*)d_in[7],  *be1 = (const float*)d_in[8];
  const float* m1 = (const float*)d_in[9],  *v1  = (const float*)d_in[10];
  const float* w2    = (const float*)d_in[11];
  const float* b2    = (const float*)d_in[12];
  const float* g2 = (const float*)d_in[13], *be2 = (const float*)d_in[14];
  const float* m2 = (const float*)d_in[15], *v2  = (const float*)d_in[16];
  const float* w3    = (const float*)d_in[17];
  const float* b3    = (const float*)d_in[18];
  float* logits = (float*)d_out;

  char* ws = (char*)d_ws;
  size_t off = 0;
  auto alloc = [&](size_t bytes) {
    char* p = ws + off;
    off = (off + bytes + 255) & ~(size_t)255;
    return p;
  };
  __bf16* feat   = (__bf16*)alloc((size_t)BATCH * SEQ * FEAT * 2);
  __bf16* wpproj = (__bf16*)alloc((size_t)HID * FEAT * 2);
  __bf16* wp1    = (__bf16*)alloc((size_t)9 * FEAT * 128 * 2);
  __bf16* wp2    = (__bf16*)alloc((size_t)9 * 128 * 64 * 2);
  __bf16* wp3    = (__bf16*)alloc((size_t)64 * 16 * 2);
  int*    winner = (int*)alloc((size_t)BATCH * HH * WW * 4);
  __bf16* act1   = (__bf16*)alloc((size_t)BATCH * HH * WW * 128 * 2);
  __bf16* fmap   = (__bf16*)alloc((size_t)BATCH * HH * WW * FEAT * 2);
  __bf16* act2   = fmap;  // fmap is dead after conv1; reuse for act2

  // Weight repacks
  pack_kn<<<(HID * FEAT + 255) / 256, 256, 0, stream>>>(pw, wpproj, HID, FEAT);
  pack_oihw<<<(FEAT * 9 * 128 + 255) / 256, 256, 0, stream>>>(w1, wp1, 128, FEAT, 9, 128);
  pack_oihw<<<(128 * 9 * 64 + 255) / 256, 256, 0, stream>>>(w2, wp2, 64, 128, 9, 64);
  pack_oihw<<<(64 * 16 + 255) / 256, 256, 0, stream>>>(w3, wp3, 9, 64, 1, 16);

  // Token projection (WMMA bf16)
  proj_wmma<<<(BATCH * SEQ / 16) * (FEAT / 16) / 8, 256, 0, stream>>>(hs, wpproj, pb, feat);

  // Rasterize winner map
  const int npix = BATCH * HH * WW;
  init_winner<<<(npix + 255) / 256, 256, 0, stream>>>(winner, npix);
  rasterize<<<(BATCH * SEQ) / 8, 256, 0, stream>>>(bbox, amask, winner);

  // Gather NHWC bf16 feature map
  gather_fmap<<<npix / 8, 256, 0, stream>>>(winner, feat, fmap);

  // Convs: async double-buffered LDS implicit GEMM on WMMA, fused BN+ReLU
  const int nblk = BATCH * (HH / 16) * (WW / 16);  // 2048 pixel tiles
  conv3x3_lds<FEAT, 2><<<nblk, 256, 0, stream>>>(
      fmap, wp1, b1, g1, be1, m1, v1, act1);
  conv3x3_lds<128, 1><<<nblk, 256, 0, stream>>>(
      act1, wp2, b2, g2, be2, m2, v2, act2);

  const int PT = BATCH * HH * (WW / 16);
  conv1x1_final<<<PT / 8, 256, 0, stream>>>(act2, wp3, b3, logits);
}
